// RefinerBlock_42348377538676
// MI455X (gfx1250) — compile-verified, matrix operand order checked by reading, not run on hardware
//
#include <hip/hip_runtime.h>
#include <hip/hip_bf16.h>
#include <math.h>

typedef __attribute__((ext_vector_type(16))) __bf16 v16bf;
typedef __attribute__((ext_vector_type(8)))  float  v8f;
typedef unsigned short u16;

// ---------- helpers ----------
__device__ __forceinline__ float gelu_exact(float x) {
    return 0.5f * x * (1.0f + erff(x * 0.70710678118654752440f));
}
__device__ __forceinline__ u16 f2bf(float f) {
    unsigned u = __float_as_uint(f);
    u += 0x7FFFu + ((u >> 16) & 1u);
    return (u16)(u >> 16);
}
// forward: which K element lane/e holds in a 16-bit 32xK fragment (ISA 7.12.2)
__device__ __forceinline__ int fragK(int lane, int e) {
    int half = lane >> 4;
    int v = e >> 1;
    int base = (v < 4) ? (2 * v) : (16 + 2 * (v - 4));
    return base + half * 8 + (e & 1);
}
// inverse: position (0..511) inside a 512-entry fragment tile for (row m, k kk)
__device__ __forceinline__ int fragPos(int m, int kk) {
    int half = (kk >> 3) & 1;
    int t = kk - 8 * half;
    int e = (t < 8) ? t : (t - 8);
    return ((m + (half << 4)) << 4) + e;
}
__device__ __forceinline__ v8f wmma_bf16(v16bf a, v16bf b, v8f c) {
    return __builtin_amdgcn_wmma_f32_16x16x32_bf16(false, a, false, b, (short)0, c, false, false);
}
__device__ __forceinline__ v16bf ldfrag(const u16* p) { return *(const v16bf*)p; }

// ---------- weight packing: row-major (Kdim x Ndim) fp32 -> bf16 B-fragment order ----------
__global__ __launch_bounds__(256) void k_pack(const float* __restrict__ W, u16* __restrict__ out,
                                              int Kdim, int Ndim) {
    int i = blockIdx.x * 256 + threadIdx.x;
    if (i >= Kdim * Ndim) return;
    int e = i & 15, lane = (i >> 4) & 31, tile = i >> 9;
    int ntn = Ndim >> 4;
    int nt = tile % ntn, kt = tile / ntn;
    int col = (nt << 4) + (lane & 15);
    int k = (kt << 5) + fragK(lane, e);
    out[i] = f2bf(W[k * Ndim + col]);
}

// ---------- LayerNorm1: one wave per row (wave32) ----------
__global__ __launch_bounds__(256) void k_ln1(const float* __restrict__ x, const float* __restrict__ g,
                                             const float* __restrict__ b, float* __restrict__ y) {
    int w = threadIdx.x >> 5, lane = threadIdx.x & 31;
    int row = blockIdx.x * 8 + w;
    const float* xr = &x[row * 256 + lane * 8];
    float4 v0 = *(const float4*)xr;
    float4 v1 = *(const float4*)(xr + 4);
    float vv[8] = {v0.x, v0.y, v0.z, v0.w, v1.x, v1.y, v1.z, v1.w};
    float s = 0.f, s2 = 0.f;
#pragma unroll
    for (int i = 0; i < 8; ++i) { s += vv[i]; s2 += vv[i] * vv[i]; }
#pragma unroll
    for (int o = 16; o > 0; o >>= 1) { s += __shfl_xor(s, o, 32); s2 += __shfl_xor(s2, o, 32); }
    float m = s * (1.f / 256.f);
    float inv = rsqrtf(s2 * (1.f / 256.f) - m * m + 1e-5f);
    int c0 = lane * 8;
#pragma unroll
    for (int i = 0; i < 8; ++i)
        y[row * 256 + c0 + i] = (vv[i] - m) * inv * g[c0 + i] + b[c0 + i];
}

// ---------- kNN: 16 nearest among 2048 (3-D), centers staged in LDS ----------
__global__ __launch_bounds__(256) void k_knn(const float* __restrict__ centers, int* __restrict__ idxw) {
    __shared__ float cx[2048], cy[2048], cz[2048], sq[2048];
    int t = threadIdx.x;
    int b = blockIdx.x >> 3;
    int n = ((blockIdx.x & 7) << 8) + t;
    for (int i = t; i < 2048; i += 256) {
        float x = centers[(b * 2048 + i) * 3 + 0];
        float y = centers[(b * 2048 + i) * 3 + 1];
        float z = centers[(b * 2048 + i) * 3 + 2];
        cx[i] = x; cy[i] = y; cz[i] = z; sq[i] = x * x + y * y + z * z;
    }
    __syncthreads();
    float qx = cx[n], qy = cy[n], qz = cz[n], qs = sq[n];
    float best[16]; int bi[16];
#pragma unroll
    for (int s = 0; s < 16; ++s) { best[s] = 3.4e38f; bi[s] = 0; }
    for (int j = 0; j < 2048; ++j) {
        float d2 = qs + sq[j] - 2.f * (qx * cx[j] + qy * cy[j] + qz * cz[j]);
        if (j == n) d2 += 1000000.0f;
        if (d2 < best[15]) {
            float dc = d2; int ic = j;
#pragma unroll
            for (int s = 0; s < 16; ++s) {
                bool sw = dc < best[s];
                float tf = best[s]; int ti = bi[s];
                if (sw) { best[s] = dc; bi[s] = ic; dc = tf; ic = ti; }
            }
        }
    }
#pragma unroll
    for (int s = 0; s < 16; ++s) idxw[(b * 2048 + n) * 16 + s] = bi[s];
}

// ---------- Message kernel: 4 queries / workgroup, M = 64 ----------
// All GEMM phases: kt-outer, 8 live accumulators (4 M-tiles x 2 N-tiles per wave)
// -> each B fragment used 4x, each A fragment used 2x.
__global__ __launch_bounds__(256) void k3_msg(const float* __restrict__ tokens, const float* __restrict__ centers,
                                              const float* __restrict__ tn, const int* __restrict__ idxw,
                                              const float* __restrict__ pw1, const float* __restrict__ pb1,
                                              const u16* __restrict__ pw2p, const float* __restrict__ pb2,
                                              const u16* __restrict__ mw1p, const float* __restrict__ mb1,
                                              const u16* __restrict__ mw2p, const float* __restrict__ mb2,
                                              float* __restrict__ tokMid) {
    extern __shared__ __align__(16) char smem[];
    u16* catF = (u16*)smem;                 // 4 mt * 24 kt * 512 = 49152 u16 (96KB)
    u16* aF   = catF + 4 * 24 * 512;        // 4 mt * 8 kt * 512  = 16384 u16 (32KB)
    float* relf = (float*)(aF + 4 * 8 * 512);  // 192 floats
    int* idxb = (int*)(relf + 192);            // 64 ints

    const int t = threadIdx.x, w = t >> 5, lane = t & 31;
    const int blk = blockIdx.x;
    const int b = blk >> 9;
    const int n0 = (blk & 511) * 4;

    if (t < 64) idxb[t] = idxw[(b * 2048 + n0 + (t >> 4)) * 16 + (t & 15)];
    __syncthreads();
    if (t < 192) {
        int q = t / 48, rem = t % 48, r = rem / 3, d = rem % 3;
        int nb = idxb[q * 16 + r];
        relf[t] = centers[(b * 2048 + nb) * 3 + d] - centers[(b * 2048 + n0 + q) * 3 + d];
    }
    __syncthreads();

    // Phase A: hpos = gelu(rel @ pw1 + pb1) -> aF (A-fragment layout, bf16)
    for (int i = t; i < 4 * 16 * 256; i += 256) {
        int c = i & 255, r = (i >> 8) & 15, q = i >> 12;
        const float* rp = &relf[(q * 16 + r) * 3];
        float v = rp[0] * pw1[c] + rp[1] * pw1[256 + c] + rp[2] * pw1[512 + c] + pb1[c];
        v = gelu_exact(v);
        aF[((q * 8 + (c >> 5)) << 9) + fragPos(r, c & 31)] = f2bf(v);
    }
    // Phase C: cat columns 0..511 (ctr_tok | nbr - ctr)
    for (int i = t; i < 4 * 16 * 512; i += 256) {
        int c = i & 511, r = (i >> 9) & 15, q = i >> 13;
        float v;
        if (c < 256) {
            v = tn[(b * 2048 + n0 + q) * 256 + c];
        } else {
            int cc = c - 256;
            v = tn[(b * 2048 + idxb[q * 16 + r]) * 256 + cc] - tn[(b * 2048 + n0 + q) * 256 + cc];
        }
        catF[((q * 24 + (c >> 5)) << 9) + fragPos(r, c & 31)] = f2bf(v);
    }
    __syncthreads();

    // Phase B: pos = hpos @ pw2 + pb2 -> cat columns 512..767
    {
        v8f acc[4][2] = {};
        for (int kt = 0; kt < 8; ++kt) {
            v16bf b0 = ldfrag(&pw2p[(((kt * 16 + w * 2 + 0) << 5) + lane) << 4]);
            v16bf b1 = ldfrag(&pw2p[(((kt * 16 + w * 2 + 1) << 5) + lane) << 4]);
#pragma unroll
            for (int mt = 0; mt < 4; ++mt) {
                v16bf a = ldfrag(&aF[(((mt * 8 + kt) << 5) + lane) << 4]);
                acc[mt][0] = wmma_bf16(a, b0, acc[mt][0]);
                acc[mt][1] = wmma_bf16(a, b1, acc[mt][1]);
            }
        }
#pragma unroll
        for (int j = 0; j < 2; ++j) {
            int nt = w * 2 + j;
            int col = nt * 16 + (lane & 15);
            float bias = pb2[col];
            int cc = 512 + col, ktc = cc >> 5, kk = cc & 31;
#pragma unroll
            for (int mt = 0; mt < 4; ++mt) {
#pragma unroll
                for (int r = 0; r < 8; ++r) {
                    int mr = r + ((lane >> 4) << 3);
                    catF[((mt * 24 + ktc) << 9) + fragPos(mr, kk)] = f2bf(acc[mt][j][r] + bias);
                }
            }
        }
    }
    __syncthreads();

    // Phase D: msg_h = gelu(cat @ mw1 + mb1) -> aF
    {
        v8f acc[4][2] = {};
        for (int kt = 0; kt < 24; ++kt) {
            v16bf b0 = ldfrag(&mw1p[(((kt * 16 + w * 2 + 0) << 5) + lane) << 4]);
            v16bf b1 = ldfrag(&mw1p[(((kt * 16 + w * 2 + 1) << 5) + lane) << 4]);
#pragma unroll
            for (int mt = 0; mt < 4; ++mt) {
                v16bf a = ldfrag(&catF[(((mt * 24 + kt) << 5) + lane) << 4]);
                acc[mt][0] = wmma_bf16(a, b0, acc[mt][0]);
                acc[mt][1] = wmma_bf16(a, b1, acc[mt][1]);
            }
        }
        __syncthreads();   // all reads of catF/aF (phase B/D inputs) complete before aF overwrite
#pragma unroll
        for (int j = 0; j < 2; ++j) {
            int nt = w * 2 + j;
            int col = nt * 16 + (lane & 15);
            float bias = mb1[col];
            int ktc = col >> 5, kk = col & 31;
#pragma unroll
            for (int mt = 0; mt < 4; ++mt) {
#pragma unroll
                for (int r = 0; r < 8; ++r) {
                    int mr = r + ((lane >> 4) << 3);
                    aF[((mt * 8 + ktc) << 9) + fragPos(mr, kk)] = f2bf(gelu_exact(acc[mt][j][r] + bias));
                }
            }
        }
    }
    __syncthreads();

    // Phase E: msg = msg_h @ mw2 + mb2; mean over K=16; residual to tokMid
    {
        v8f acc[4][2] = {};
        for (int kt = 0; kt < 8; ++kt) {
            v16bf b0 = ldfrag(&mw2p[(((kt * 16 + w * 2 + 0) << 5) + lane) << 4]);
            v16bf b1 = ldfrag(&mw2p[(((kt * 16 + w * 2 + 1) << 5) + lane) << 4]);
#pragma unroll
            for (int mt = 0; mt < 4; ++mt) {
                v16bf a = ldfrag(&aF[(((mt * 8 + kt) << 5) + lane) << 4]);
                acc[mt][0] = wmma_bf16(a, b0, acc[mt][0]);
                acc[mt][1] = wmma_bf16(a, b1, acc[mt][1]);
            }
        }
#pragma unroll
        for (int j = 0; j < 2; ++j) {
            int nt = w * 2 + j;
#pragma unroll
            for (int mt = 0; mt < 4; ++mt) {
                v8f a8 = acc[mt][j];
                float s = a8[0] + a8[1] + a8[2] + a8[3] + a8[4] + a8[5] + a8[6] + a8[7];
                s += __shfl_xor(s, 16, 32);
                if (lane < 16) {
                    int col = nt * 16 + lane;
                    int row = b * 2048 + n0 + mt;
                    tokMid[row * 256 + col] = tokens[row * 256 + col] + mb2[col] + s * 0.0625f;
                }
            }
        }
    }
}

// ---------- FFN part 1: LN2 + gelu(t2 @ fw1 + fb1), hidden stored packed in A-frag order ----------
__global__ __launch_bounds__(256) void k4a_ffn1(const float* __restrict__ tokMid, const float* __restrict__ g,
                                                const float* __restrict__ bv, const u16* __restrict__ fw1p,
                                                const float* __restrict__ fb1, u16* __restrict__ hp) {
    extern __shared__ __align__(16) char smem[];
    u16* t2F = (u16*)smem;             // 4*8*512 u16 (32KB)
    u16* stag = t2F + 4 * 8 * 512;     // 8 waves * 4 frags * 512 u16 (32KB)
    float* psum = (float*)(stag + 8 * 4 * 512);
    float* psq  = psum + 256;
    float* mrw  = psq + 256;
    float* irw  = mrw + 64;

    const int t = threadIdx.x, w = t >> 5, lane = t & 31;
    const int blk = blockIdx.x;
    const int row0 = blk * 64;

    {
        int r = t >> 2, qq = t & 3;
        const float* x = &tokMid[(row0 + r) * 256 + qq * 64];
        float s = 0.f, s2 = 0.f;
        for (int i = 0; i < 64; i += 4) {
            float4 v4 = *(const float4*)(x + i);
            s += v4.x + v4.y + v4.z + v4.w;
            s2 += v4.x * v4.x + v4.y * v4.y + v4.z * v4.z + v4.w * v4.w;
        }
        psum[t] = s; psq[t] = s2;
    }
    __syncthreads();
    if (t < 64) {
        float s  = psum[t * 4] + psum[t * 4 + 1] + psum[t * 4 + 2] + psum[t * 4 + 3];
        float s2 = psq[t * 4] + psq[t * 4 + 1] + psq[t * 4 + 2] + psq[t * 4 + 3];
        float m = s * (1.f / 256.f);
        mrw[t] = m;
        irw[t] = rsqrtf(s2 * (1.f / 256.f) - m * m + 1e-5f);
    }
    __syncthreads();
    for (int i = t; i < 64 * 256; i += 256) {
        int c = i & 255, r = i >> 8;
        float x = tokMid[(row0 + r) * 256 + c];
        float v = (x - mrw[r]) * irw[r] * g[c] + bv[c];
        t2F[(((r >> 4) * 8 + (c >> 5)) << 9) + fragPos(r & 15, c & 31)] = f2bf(v);
    }
    __syncthreads();

    u16* st = &stag[(w * 4) << 9];     // 4 fragments for this wave (one per mt)
    for (int fi = 0; fi < 4; ++fi) {
        int ntA = w + (fi << 3);       // 0..31 across waves
        v8f acc[4][2] = {};
        for (int kt = 0; kt < 8; ++kt) {
            v16bf b0 = ldfrag(&fw1p[(((kt * 64 + ntA * 2 + 0) << 5) + lane) << 4]);
            v16bf b1 = ldfrag(&fw1p[(((kt * 64 + ntA * 2 + 1) << 5) + lane) << 4]);
#pragma unroll
            for (int mt = 0; mt < 4; ++mt) {
                v16bf a = ldfrag(&t2F[(((mt * 8 + kt) << 5) + lane) << 4]);
                acc[mt][0] = wmma_bf16(a, b0, acc[mt][0]);
                acc[mt][1] = wmma_bf16(a, b1, acc[mt][1]);
            }
        }
#pragma unroll
        for (int j = 0; j < 2; ++j) {
            int nt = ntA * 2 + j;
            int col = nt * 16 + (lane & 15);
            float bias = fb1[col];
            int kk = (j << 4) + (lane & 15);
#pragma unroll
            for (int mt = 0; mt < 4; ++mt) {
#pragma unroll
                for (int r = 0; r < 8; ++r) {
                    int mr = r + ((lane >> 4) << 3);
                    st[(mt << 9) + fragPos(mr, kk)] = f2bf(gelu_exact(acc[mt][j][r] + bias));
                }
            }
        }
#pragma unroll
        for (int mt = 0; mt < 4; ++mt) {
            v16bf vv = *(const v16bf*)&st[(mt << 9) + (lane << 4)];
            *(v16bf*)&hp[((((blk * 4 + mt) * 32 + ntA) << 5) + lane) << 4] = vv;
        }
    }
}

// ---------- FFN part 2: out = tokMid + h @ fw2 + fb2 ----------
__global__ __launch_bounds__(256) void k4b_ffn2(const float* __restrict__ tokMid, const u16* __restrict__ hp,
                                                const u16* __restrict__ fw2p, const float* __restrict__ fb2,
                                                float* __restrict__ out) {
    const int t = threadIdx.x, w = t >> 5, lane = t & 31;
    const int blk = blockIdx.x;
    const int row0 = blk * 64;
    v8f acc[4][2] = {};
    for (int kt = 0; kt < 32; ++kt) {
        v16bf b0 = ldfrag(&fw2p[(((kt * 16 + w * 2 + 0) << 5) + lane) << 4]);
        v16bf b1 = ldfrag(&fw2p[(((kt * 16 + w * 2 + 1) << 5) + lane) << 4]);
#pragma unroll
        for (int mt = 0; mt < 4; ++mt) {
            v16bf a = ldfrag(&hp[((((blk * 4 + mt) * 32 + kt) << 5) + lane) << 4]);
            acc[mt][0] = wmma_bf16(a, b0, acc[mt][0]);
            acc[mt][1] = wmma_bf16(a, b1, acc[mt][1]);
        }
    }
#pragma unroll
    for (int j = 0; j < 2; ++j) {
        int nt = w * 2 + j;
        int col = nt * 16 + (lane & 15);
        float bias = fb2[col];
#pragma unroll
        for (int mt = 0; mt < 4; ++mt) {
#pragma unroll
            for (int r = 0; r < 8; ++r) {
                int row = row0 + mt * 16 + r + ((lane >> 4) << 3);
                out[row * 256 + col] = tokMid[row * 256 + col] + bias + acc[mt][j][r];
            }
        }
    }
}

// ---------- host ----------
extern "C" void kernel_launch(void* const* d_in, const int* in_sizes, int n_in,
                              void* d_out, int out_size, void* d_ws, size_t ws_size,
                              hipStream_t stream) {
    const float* tokens  = (const float*)d_in[0];
    const float* centers = (const float*)d_in[1];
    const float* n1_g = (const float*)d_in[2];
    const float* n1_b = (const float*)d_in[3];
    const float* pw1  = (const float*)d_in[4];
    const float* pb1  = (const float*)d_in[5];
    const float* pw2  = (const float*)d_in[6];
    const float* pb2  = (const float*)d_in[7];
    const float* mw1  = (const float*)d_in[8];
    const float* mb1  = (const float*)d_in[9];
    const float* mw2  = (const float*)d_in[10];
    const float* mb2  = (const float*)d_in[11];
    const float* n2_g = (const float*)d_in[12];
    const float* n2_b = (const float*)d_in[13];
    const float* fw1  = (const float*)d_in[14];
    const float* fb1  = (const float*)d_in[15];
    const float* fw2  = (const float*)d_in[16];
    const float* fb2  = (const float*)d_in[17];
    float* out = (float*)d_out;

    char* wsb = (char*)d_ws;
    size_t off = 0;
    auto alloc = [&](size_t bytes) { char* p = wsb + off; off = (off + bytes + 255) & ~(size_t)255; return p; };
    float* tn     = (float*)alloc(8192ull * 256 * 4);
    float* tokMid = (float*)alloc(8192ull * 256 * 4);
    int*   idxw   = (int*)  alloc(8192ull * 16 * 4);
    u16*   pw2p   = (u16*)  alloc(256ull * 256 * 2);
    u16*   mw1p   = (u16*)  alloc(768ull * 256 * 2);
    u16*   mw2p   = (u16*)  alloc(256ull * 256 * 2);
    u16*   fw1p   = (u16*)  alloc(256ull * 1024 * 2);
    u16*   fw2p   = (u16*)  alloc(1024ull * 256 * 2);
    u16*   hp     = (u16*)  alloc(8192ull * 1024 * 2);
    (void)ws_size; (void)n_in; (void)in_sizes; (void)out_size;

    const int SM3  = (4 * 24 * 512 + 4 * 8 * 512) * 2 + 192 * 4 + 64 * 4;
    const int SM4A = (4 * 8 * 512 + 8 * 4 * 512) * 2 + 640 * 4;
    hipFuncSetAttribute((const void*)k3_msg, hipFuncAttributeMaxDynamicSharedMemorySize, SM3);
    hipFuncSetAttribute((const void*)k4a_ffn1, hipFuncAttributeMaxDynamicSharedMemorySize, SM4A);

    k_pack<<<(256 * 256) / 256, 256, 0, stream>>>(pw2, pw2p, 256, 256);
    k_pack<<<(768 * 256) / 256, 256, 0, stream>>>(mw1, mw1p, 768, 256);
    k_pack<<<(256 * 256) / 256, 256, 0, stream>>>(mw2, mw2p, 256, 256);
    k_pack<<<(256 * 1024) / 256, 256, 0, stream>>>(fw1, fw1p, 256, 1024);
    k_pack<<<(1024 * 256) / 256, 256, 0, stream>>>(fw2, fw2p, 1024, 256);

    k_ln1<<<1024, 256, 0, stream>>>(tokens, n1_g, n1_b, tn);
    k_knn<<<32, 256, 0, stream>>>(centers, idxw);

    k3_msg<<<2048, 256, SM3, stream>>>(tokens, centers, tn, idxw, pw1, pb1,
                                       pw2p, pb2, mw1p, mb1, mw2p, mb2, tokMid);

    k4a_ffn1<<<128, 256, SM4A, stream>>>(tokMid, n2_g, n2_b, fw1p, fb1, hp);
    k4b_ffn2<<<128, 256, 0, stream>>>(tokMid, hp, fw2p, fb2, out);
}
